// MyModel_61933428412641
// MI455X (gfx1250) — matrix operand order repaired and checked by reading, not compiled, not run on hardware
//
#include <hip/hip_runtime.h>
#include <hip/hip_bf16.h>

typedef __bf16 bf16;
typedef __attribute__((ext_vector_type(16))) __bf16 v16bf;
typedef __attribute__((ext_vector_type(8)))  float  v8f;

#define B_DIM 256
#define T_DIM 512
#define I_DIM 300
#define I_PAD 320
#define H_DIM 128
#define G4    512   // 4*H

union Frag16 { v16bf v; uint4 q[2]; };

// A-matrix fragment (16x32 bf16) from row-major [16 x stride] tile.
// lane L: row = L&15; chunks of 8 K-values at k = kk+(L>>4)*8 and +16.
__device__ inline v16bf load_a_frag(const bf16* base, int stride, int kk, int lane) {
    const int r  = lane & 15;
    const int kh = (lane >> 4) * 8;
    const bf16* p = base + r * stride + kk + kh;
    Frag16 f;
    f.q[0] = *(const uint4*)(p);
    f.q[1] = *(const uint4*)(p + 16);
    return f.v;
}

// B-matrix fragment (32x16 bf16) from row-major [16(N) x stride(K)] tile (i.e. B^T layout).
// lane L: col n = L&15; chunks of 8 K-values at k = kk+(L>>4)*8 and +16.
__device__ inline v16bf load_b_frag(const bf16* base, int stride, int kk, int lane) {
    const int n  = lane & 15;
    const int kh = (lane >> 4) * 8;
    const bf16* p = base + n * stride + kk + kh;
    Frag16 f;
    f.q[0] = *(const uint4*)(p);
    f.q[1] = *(const uint4*)(p + 16);
    return f.v;
}

__device__ inline float fsig(float x)  { return 1.0f / (1.0f + __expf(-x)); }
__device__ inline float ftanh(float x) { float e = __expf(-2.0f * x); return (1.0f - e) / (1.0f + e); }

// ---------------- Kernel 1: weight prep (fp32 -> bf16, pad K, fuse biases) ----------------
__global__ __launch_bounds__(256) void prep_weights(
        const float* __restrict__ Wih, const float* __restrict__ Whh,
        const float* __restrict__ bih, const float* __restrict__ bhh,
        bf16* __restrict__ WihB, bf16* __restrict__ WhhB, float* __restrict__ bias) {
    const int i0 = blockIdx.x * 256 + threadIdx.x;
    const int stride = gridDim.x * 256;
    for (int i = i0; i < G4 * I_PAD; i += stride) {
        int n = i / I_PAD, k = i - n * I_PAD;
        WihB[i] = (bf16)((k < I_DIM) ? Wih[n * I_DIM + k] : 0.0f);
    }
    for (int i = i0; i < G4 * H_DIM; i += stride) WhhB[i] = (bf16)Whh[i];
    for (int i = i0; i < G4; i += stride) bias[i] = bih[i] + bhh[i];
}

// ---------------- Kernel 2: xg = x @ W_ih^T + (b_ih+b_hh)  [131072 x 512] ----------------
// Block tile: 32(M) x 256(N); 8 waves = 2(M) x 4(N); each wave does a 16(M) x 64(N) strip
// (4 accumulators, A-frag reused 4x). K staged in LDS as bf16, zero-padded 300 -> 320.
__global__ __launch_bounds__(256) void gemm_xg(
        const float* __restrict__ x, const bf16* __restrict__ WihB,
        const float* __restrict__ bias, float* __restrict__ xg) {
    __shared__ bf16 As[32 * I_PAD];  // 20 KB
    const int tid   = threadIdx.x;
    const int mbase = blockIdx.x * 32;
    const int nbase = blockIdx.y * 256;

    // Stage A tile: 32 rows of 300 f32 (75 float4 each; 1200B rows stay 16B aligned) -> bf16
    const float4* x4 = (const float4*)x;
    for (int idx = tid; idx < 32 * 75; idx += 256) {
        int r = idx / 75, c = idx - r * 75;
        float4 v = x4[(size_t)(mbase + r) * 75 + c];
        int o = r * I_PAD + c * 4;
        As[o + 0] = (bf16)v.x; As[o + 1] = (bf16)v.y;
        As[o + 2] = (bf16)v.z; As[o + 3] = (bf16)v.w;
    }
    for (int idx = tid; idx < 32 * 20; idx += 256) {  // zero pad K = 300..319
        int r = idx / 20, c = idx - r * 20;
        As[r * I_PAD + I_DIM + c] = (bf16)0.0f;
    }
    __syncthreads();

    const int lane = tid & 31, wid = tid >> 5;
    const int mw   = (wid & 1) * 16;        // 0 or 16
    const int nw0  = nbase + (wid >> 1) * 64;  // 64-col strip per wave
    const int ncl  = lane & 15;

    v8f acc[4];
    #pragma unroll
    for (int j = 0; j < 4; ++j) {
        const float bv = bias[nw0 + j * 16 + ncl];
        #pragma unroll
        for (int i = 0; i < 8; ++i) acc[j][i] = bv;
    }

    #pragma unroll
    for (int kk = 0; kk < I_PAD; kk += 32) {
        v16bf a = load_a_frag(As + mw * I_PAD, I_PAD, kk, lane);
        #pragma unroll
        for (int j = 0; j < 4; ++j) {
            v16bf b = load_b_frag(WihB + (size_t)(nw0 + j * 16) * I_PAD, I_PAD, kk, lane);
            acc[j] = __builtin_amdgcn_wmma_f32_16x16x32_bf16(false, a, false, b, (short)0,
                                                             acc[j], false, false);
        }
    }

    const int mrow0 = mbase + mw + (lane >> 4) * 8;
    #pragma unroll
    for (int j = 0; j < 4; ++j)
        #pragma unroll
        for (int rr = 0; rr < 8; ++rr)
            xg[(size_t)(mrow0 + rr) * G4 + (nw0 + j * 16 + ncl)] = acc[j][rr];
}

// ---------------- Kernel 3: recurrent scan ----------------
// 16 blocks x 256 thr (8 waves). Block owns 16 batch rows for all T.
// Wave w owns h-columns [16w,16w+16): gate tiles at cols 16w, +128, +256, +384.
// h double-buffered in LDS (bf16) -> single barrier per timestep.
__global__ __launch_bounds__(256) void lstm_scan(
        const float* __restrict__ xg, const bf16* __restrict__ WhhB,
        const float* __restrict__ fcw, const float* __restrict__ fcb,
        float* __restrict__ out) {
    __shared__ bf16  hB[2 * 16 * H_DIM];  // double-buffered h (WMMA A operand)
    __shared__ float hF[16 * H_DIM];      // f32 h, written once after the loop
    const int tid   = threadIdx.x;
    const int lane  = tid & 31, wid = tid >> 5;
    const int bbase = blockIdx.x * 16;
    const int jbase = wid * 16;

    for (int i = tid; i < 16 * H_DIM; i += 256) hB[i] = (bf16)0.0f;  // buffer 0 = h(-1) = 0
    __syncthreads();

    // Preload W_hh B-fragments (t-invariant): [k-step][gate]
    v16bf bfr[4][4];
    #pragma unroll
    for (int ks = 0; ks < 4; ++ks)
        #pragma unroll
        for (int g = 0; g < 4; ++g)
            bfr[ks][g] = load_b_frag(WhhB + (size_t)(g * H_DIM + jbase) * H_DIM,
                                     H_DIM, ks * 32, lane);

    const int mloc = (lane >> 4) * 8;
    const int ncl  = lane & 15;
    float cst[8], hlast[8];
    #pragma unroll
    for (int i = 0; i < 8; ++i) { cst[i] = 0.0f; hlast[i] = 0.0f; }

    int p = 0;  // buffer holding h(t-1)
    for (int t = 0; t < T_DIM; ++t) {
        // Init accumulators with precomputed input projections (biases fused),
        // and prefetch next timestep's xg while WMMAs + transcendentals run.
        v8f acc[4];
        #pragma unroll
        for (int g = 0; g < 4; ++g) {
            const int col = g * H_DIM + jbase + ncl;
            #pragma unroll
            for (int rr = 0; rr < 8; ++rr) {
                const float* pp = &xg[((size_t)(bbase + mloc + rr) * T_DIM + t) * G4 + col];
                acc[g][rr] = *pp;
                if (t + 1 < T_DIM) __builtin_prefetch(pp + G4, 0, 3);
            }
        }
        // gates += h(t-1) @ W_hh^T   (16 WMMAs per wave per step)
        const bf16* hrd = hB + p * (16 * H_DIM);
        #pragma unroll
        for (int ks = 0; ks < 4; ++ks) {
            v16bf a = load_a_frag(hrd, H_DIM, ks * 32, lane);
            #pragma unroll
            for (int g = 0; g < 4; ++g)
                acc[g] = __builtin_amdgcn_wmma_f32_16x16x32_bf16(false, a, false, bfr[ks][g],
                                                                 (short)0, acc[g], false, false);
        }

        bf16* hwr = hB + (p ^ 1) * (16 * H_DIM);
        #pragma unroll
        for (int rr = 0; rr < 8; ++rr) {
            float iv = fsig(acc[0][rr]);
            float fv = fsig(acc[1][rr]);
            float gv = ftanh(acc[2][rr]);
            float ov = fsig(acc[3][rr]);
            float cn = fv * cst[rr] + iv * gv;
            cst[rr] = cn;
            float hv = ov * ftanh(cn);
            hlast[rr] = hv;
            hwr[(mloc + rr) * H_DIM + jbase + ncl] = (bf16)hv;
        }
        __syncthreads();  // new h visible; old-buffer reads retired
        p ^= 1;
    }

    // Final FC: out[b] = h_last[b,:] . fc_w + fc_b  (f32 h via LDS shadow)
    #pragma unroll
    for (int rr = 0; rr < 8; ++rr)
        hF[(mloc + rr) * H_DIM + jbase + ncl] = hlast[rr];
    __syncthreads();
    if (tid < 16) {
        float s = fcb[0];
        for (int j = 0; j < H_DIM; ++j) s += hF[tid * H_DIM + j] * fcw[j];
        out[bbase + tid] = s;
    }
}

extern "C" void kernel_launch(void* const* d_in, const int* in_sizes, int n_in,
                              void* d_out, int out_size, void* d_ws, size_t ws_size,
                              hipStream_t stream) {
    const float* x    = (const float*)d_in[0];
    const float* Wih  = (const float*)d_in[1];
    const float* Whh  = (const float*)d_in[2];
    const float* bih  = (const float*)d_in[3];
    const float* bhh  = (const float*)d_in[4];
    const float* fcw  = (const float*)d_in[5];
    const float* fcb  = (const float*)d_in[6];
    float* out = (float*)d_out;

    char* ws = (char*)d_ws;
    const size_t XG_BYTES   = (size_t)B_DIM * T_DIM * G4 * sizeof(float);   // 268 MB
    const size_t WIHB_BYTES = (size_t)G4 * I_PAD * sizeof(bf16);
    const size_t WHHB_BYTES = (size_t)G4 * H_DIM * sizeof(bf16);
    float* xg   = (float*)ws;
    bf16*  WihB = (bf16*)(ws + XG_BYTES);
    bf16*  WhhB = (bf16*)(ws + XG_BYTES + WIHB_BYTES);
    float* bias = (float*)(ws + XG_BYTES + WIHB_BYTES + WHHB_BYTES);

    prep_weights<<<dim3(64), dim3(256), 0, stream>>>(Wih, Whh, bih, bhh, WihB, WhhB, bias);
    gemm_xg<<<dim3((B_DIM * T_DIM) / 32, G4 / 256), dim3(256), 0, stream>>>(x, WihB, bias, xg);
    lstm_scan<<<dim3(B_DIM / 16), dim3(256), 0, stream>>>(xg, WhhB, fcw, fcb, out);
}
